// MccBlock_4930622455816
// MI455X (gfx1250) — compile-verified, matrix operand order checked by reading, not run on hardware
//
#include <hip/hip_runtime.h>
#include <hip/hip_bf16.h>
#include <hip/hip_fp16.h>

// ---------------- problem constants (from reference) ----------------
#define D_MODEL 64
#define D_INNER 128
#define D_STATE 16
#define NB      8
#define HW      64
#define LSEQ    4096              // 64*64
#define NTOK    (NB * LSEQ)       // 32768 tokens
#define EPSBN   1e-5f

#define USE_ASYNC_LDS 1           // global_load_async_to_lds_b128 path in k_x2t

typedef __attribute__((ext_vector_type(16))) _Float16 v16h;
typedef __attribute__((ext_vector_type(8)))  _Float16 v8h;
typedef __attribute__((ext_vector_type(8)))  float    v8f;

__device__ __forceinline__ float siluf(float x)     { return x / (1.f + __expf(-x)); }
__device__ __forceinline__ float softplusf(float x) { return (x > 20.f) ? x : log1pf(__expf(x)); }

__device__ __forceinline__ v8f wmma16(v16h a, v16h b, v8f c) {
  return __builtin_amdgcn_wmma_f32_16x16x32_f16(false, a, false, b, (short)0, c, false, false);
}

// ---- fragment loaders -------------------------------------------------
// CDNA5 16-bit operand layout: element e<8 -> K = 8*hl + e ; e>=8 -> K = 16 + 8*hl + (e-8).
// From an f16 K-contiguous row at [kc + 8*hl]: lo half there, hi half +16.
__device__ __forceinline__ v16h fragh(const _Float16* p) {
  const v8h lo = *(const v8h*)p;
  const v8h hi = *(const v8h*)(p + 16);
  return __builtin_shufflevector(lo, hi, 0,1,2,3,4,5,6,7,8,9,10,11,12,13,14,15);
}
__device__ __forceinline__ v16h pack16(float4 a0, float4 a1, float4 a2, float4 a3) {
  v16h r;
  r[0]=(_Float16)a0.x;  r[1]=(_Float16)a0.y;  r[2]=(_Float16)a0.z;  r[3]=(_Float16)a0.w;
  r[4]=(_Float16)a1.x;  r[5]=(_Float16)a1.y;  r[6]=(_Float16)a1.z;  r[7]=(_Float16)a1.w;
  r[8]=(_Float16)a2.x;  r[9]=(_Float16)a2.y;  r[10]=(_Float16)a2.z; r[11]=(_Float16)a2.w;
  r[12]=(_Float16)a3.x; r[13]=(_Float16)a3.y; r[14]=(_Float16)a3.z; r[15]=(_Float16)a3.w;
  return r;
}

// ======================================================================
// 0a) transpose x (B,C,L) -> xt16 (B,L,C) f16; tile staged via async DMA
// ======================================================================
__global__ void k_x2t(const float* __restrict__ x, _Float16* __restrict__ xt) {
  __shared__ __align__(16) float s[64 * 68];          // 68-pad: keeps 16B alignment
  const int tid = threadIdx.x;
  const int b = blockIdx.x >> 6, lt = blockIdx.x & 63;
#if USE_ASYNC_LDS
  // stage 64x64 f32 tile with ASYNCcnt-tracked global->LDS DMA (no VGPR round-trip)
  for (int q = 0; q < 4; ++q) {
    const int v4 = q * 256 + tid;                     // float4 index 0..1023
    const int c = v4 >> 4, l4 = (v4 & 15) * 4;
    const unsigned long long ga =
        (unsigned long long)&x[((size_t)b * D_MODEL + c) * LSEQ + lt * 64 + l4];
    const unsigned lp = (unsigned)(size_t)&s[c * 68 + l4];
    asm volatile("global_load_async_to_lds_b128 %0, %1, off"
                 :: "v"(lp), "v"(ga) : "memory");
  }
  asm volatile("s_wait_asynccnt 0x0" ::: "memory");
#else
  for (int q = 0; q < 4; ++q) {
    const int v4 = q * 256 + tid;
    const int c = v4 >> 4, l4 = (v4 & 15) * 4;
    *(float4*)&s[c * 68 + l4] =
        *(const float4*)&x[((size_t)b * D_MODEL + c) * LSEQ + lt * 64 + l4];
  }
#endif
  __syncthreads();
  for (int idx = tid; idx < 4096; idx += 256) {
    const int l0 = idx >> 6, c = idx & 63;
    xt[((size_t)b * LSEQ + lt * 64 + l0) * D_MODEL + c] = (_Float16)s[c * 68 + l0];
  }
}

// ======================================================================
// 0b) weight preps: plain f32->f16 copy; conv OIHW -> [s][co][ci];
//     x_proj padded to 48 rows (zeros)
// ======================================================================
__global__ void k_f32tof16(const float* __restrict__ src, _Float16* __restrict__ dst, int n) {
  const int i = blockIdx.x * 256 + threadIdx.x;
  if (i < n) dst[i] = (_Float16)src[i];
}
__global__ void k_wprep(const float* __restrict__ w, _Float16* __restrict__ wt) {
  const int idx = blockIdx.x * 256 + threadIdx.x;
  if (idx >= 9 * 64 * 64) return;
  const int s = idx >> 12, rem = idx & 4095;
  const int co = rem >> 6, ci = rem & 63;
  wt[idx] = (_Float16)w[(size_t)(co * 64 + ci) * 9 + s];
}
__global__ void k_xpw(const float* __restrict__ src, _Float16* __restrict__ dst) {
  const int i = blockIdx.x * 256 + threadIdx.x;       // 48*128
  if (i >= 48 * 128) return;
  const int n = i >> 7, k = i & 127;
  dst[i] = (n < 36) ? (_Float16)src[n * 128 + k] : (_Float16)0.f;
}

// ======================================================================
// 1) in_proj: (32768 x 64) @ (64 x 256)^T -> xm | z   (all-f16 operands)
// ======================================================================
__global__ void k_inproj(const _Float16* __restrict__ xt, const _Float16* __restrict__ w,
                         float* __restrict__ xm, float* __restrict__ z) {
  const int lane = threadIdx.x, hl = lane >> 4, m = lane & 15;
  const int t0 = blockIdx.x * 16;
  const int nt = blockIdx.y * 8 + threadIdx.y;       // 0..15
  const int n  = nt * 16 + m;
  const _Float16* arow = xt + (size_t)(t0 + m) * D_MODEL;
  const _Float16* brow = w + n * D_MODEL;
  v8f acc = {};
#pragma unroll
  for (int kc = 0; kc < D_MODEL; kc += 32)
    acc = wmma16(fragh(arow + kc + 8 * hl), fragh(brow + kc + 8 * hl), acc);
  float* dst = (nt < 8) ? (xm + n) : (z + (n - D_INNER));   // wave-uniform
#pragma unroll
  for (int r = 0; r < 8; ++r)
    dst[(size_t)(t0 + r + 8 * hl) * D_INNER] = acc[r];
}

// ======================================================================
// 2) causal depthwise conv1d (k=4) + SiLU; writes f32 (scan) + f16 (GEMM A)
// ======================================================================
__global__ void k_dwconv_silu(const float* __restrict__ xm, const float* __restrict__ wd,
                              const float* __restrict__ bd, float* __restrict__ xs,
                              _Float16* __restrict__ xs16) {
  const int id = blockIdx.x * blockDim.x + threadIdx.x;
  if (id >= NTOK * D_INNER) return;
  const int d = id & (D_INNER - 1);
  const int t = id >> 7;
  const int b = t / LSEQ, l = t % LSEQ;
  float acc = bd[d];
#pragma unroll
  for (int k = 0; k < 4; ++k) {
    const int ls = l - 3 + k;
    if (ls >= 0) acc += xm[((size_t)b * LSEQ + ls) * D_INNER + d] * wd[d * 4 + k];
  }
  const float v = siluf(acc);
  xs[id] = v;
  xs16[id] = (_Float16)v;
}

// ======================================================================
// 3) x_proj: (32768 x 128) @ (128 x 48)^T (rows 36..47 pre-zeroed)
// ======================================================================
__global__ void k_xproj(const _Float16* __restrict__ xs16, const _Float16* __restrict__ w,
                        float* __restrict__ xdbl) {
  const int lane = threadIdx.x, hl = lane >> 4, m = lane & 15;
  const int t0 = blockIdx.x * 16;
  const int n  = threadIdx.y * 16 + m;   // 0..47
  const _Float16* arow = xs16 + (size_t)(t0 + m) * D_INNER;
  const _Float16* brow = w + n * D_INNER;
  v8f acc = {};
#pragma unroll
  for (int kc = 0; kc < D_INNER; kc += 32)
    acc = wmma16(fragh(arow + kc + 8 * hl), fragh(brow + kc + 8 * hl), acc);
#pragma unroll
  for (int r = 0; r < 8; ++r) {
    const int t = t0 + r + 8 * hl;
    if (n < 36) xdbl[(size_t)t * 36 + n] = acc[r];
  }
}

// ======================================================================
// 4) selective scan (fused dt proj + softplus + D skip); 16 lanes = 16 states
// ======================================================================
__global__ void k_scan(const float* __restrict__ xdbl, const float* __restrict__ xs,
                       const float* __restrict__ dtw, const float* __restrict__ dtb,
                       const float* __restrict__ A_log, const float* __restrict__ Dp,
                       float* __restrict__ y) {
  const int lane = threadIdx.x, hl = lane >> 4, n = lane & 15;
  const int wg = blockIdx.x * 8 + threadIdx.y;     // 0..511
  const int pi = wg * 2 + hl;                      // (b,d) pair 0..1023
  const int b = pi / D_INNER, d = pi % D_INNER;
  const float a_dn = -__expf(A_log[d * D_STATE + n]);
  const float Dd = Dp[d];
  const float w0 = dtw[d * 4 + 0], w1 = dtw[d * 4 + 1];
  const float w2 = dtw[d * 4 + 2], w3 = dtw[d * 4 + 3];
  const float bias = dtb[d];
  float h = 0.f;
  for (int l = 0; l < LSEQ; ++l) {
    const size_t t = (size_t)b * LSEQ + l;
    const float* xb = xdbl + t * 36;
    __builtin_prefetch(xb + 36 * 8, 0, 0);                       // global_prefetch_b8
    __builtin_prefetch(xs + t * D_INNER + 8 * D_INNER + d, 0, 0);
    const float dt = softplusf(xb[0] * w0 + xb[1] * w1 + xb[2] * w2 + xb[3] * w3 + bias);
    const float Bn = xb[4 + n], Cn = xb[20 + n];
    const float s  = xs[t * D_INNER + d];
    h = __expf(dt * a_dn) * h + dt * Bn * s;
    float p = h * Cn;
    p += __shfl_xor(p, 1, 16);
    p += __shfl_xor(p, 2, 16);
    p += __shfl_xor(p, 4, 16);
    p += __shfl_xor(p, 8, 16);
    if (n == 0) y[t * D_INNER + d] = p + s * Dd;
  }
}

// ======================================================================
// 5) gate + out_proj + PReLU; writes conv1 input in f16 channel-major
//    cin16[b][i=w][j=h][c]
// ======================================================================
__global__ void k_gate_outproj(const float* __restrict__ y, const float* __restrict__ z,
                               const _Float16* __restrict__ w, const float* __restrict__ pr,
                               _Float16* __restrict__ cin16) {
  const int lane = threadIdx.x, hl = lane >> 4, m = lane & 15;
  const int t0 = blockIdx.x * 16;
  const int n  = threadIdx.y * 16 + m;   // 0..63
  const float* yrow = y + (size_t)(t0 + m) * D_INNER;
  const float* zrow = z + (size_t)(t0 + m) * D_INNER;
  const _Float16* brow = w + n * D_INNER;
  v8f acc = {};
#pragma unroll
  for (int kc = 0; kc < D_INNER; kc += 32) {
    const int o = kc + 8 * hl;
    float4 y0 = *(const float4*)&yrow[o],      y1 = *(const float4*)&yrow[o + 4];
    float4 y2 = *(const float4*)&yrow[o + 16], y3 = *(const float4*)&yrow[o + 20];
    float4 z0 = *(const float4*)&zrow[o],      z1 = *(const float4*)&zrow[o + 4];
    float4 z2 = *(const float4*)&zrow[o + 16], z3 = *(const float4*)&zrow[o + 20];
    y0.x *= siluf(z0.x); y0.y *= siluf(z0.y); y0.z *= siluf(z0.z); y0.w *= siluf(z0.w);
    y1.x *= siluf(z1.x); y1.y *= siluf(z1.y); y1.z *= siluf(z1.z); y1.w *= siluf(z1.w);
    y2.x *= siluf(z2.x); y2.y *= siluf(z2.y); y2.z *= siluf(z2.z); y2.w *= siluf(z2.w);
    y3.x *= siluf(z3.x); y3.y *= siluf(z3.y); y3.z *= siluf(z3.z); y3.w *= siluf(z3.w);
    acc = wmma16(pack16(y0, y1, y2, y3), fragh(brow + o), acc);
  }
  const float al = pr[0];
#pragma unroll
  for (int r = 0; r < 8; ++r) {
    const int t = t0 + r + 8 * hl;
    const int b = t / LSEQ, l = t % LSEQ;
    const int hh = l >> 6, ww = l & 63;    // l = h*64 + w ; conv input (i=w, j=h)
    float v = acc[r];
    v = (v >= 0.f) ? v : al * v;
    cin16[(((size_t)b * HW + ww) * HW + hh) * D_MODEL + n] = (_Float16)v;
  }
}

// ======================================================================
// 6) residual branch: 1x1 conv + BN (all-f16 operands)
// ======================================================================
__global__ void k_res(const _Float16* __restrict__ xt, const _Float16* __restrict__ w,
                      const float* __restrict__ bn, float* __restrict__ out) {
  const int lane = threadIdx.x, hl = lane >> 4, m = lane & 15;
  const int bx = blockIdx.x;               // 8*256
  const int b = bx >> 8, pt = bx & 255;
  const int p0 = pt * 16;
  const int n  = threadIdx.y * 16 + m;     // co
  const _Float16* arow = xt + (size_t)(b * LSEQ + p0 + m) * D_MODEL;
  const _Float16* brow = w + n * D_MODEL;
  v8f acc = {};
#pragma unroll
  for (int kc = 0; kc < D_MODEL; kc += 32)
    acc = wmma16(fragh(arow + kc + 8 * hl), fragh(brow + kc + 8 * hl), acc);
  const float g = bn[n], be = bn[64 + n], mu = bn[128 + n], var = bn[192 + n];
  const float sc = g * rsqrtf(var + EPSBN);
#pragma unroll
  for (int r = 0; r < 8; ++r) {
    const int p = p0 + r + 8 * hl;
    out[((size_t)b * D_MODEL + n) * LSEQ + p] = (acc[r] - mu) * sc + be;
  }
}

// ======================================================================
// 7) 3x3 conv + BN + PReLU as 9 shifted channel-GEMMs; f16 channel-major
//    operands from global (L2-resident), 18 WMMAs/wave.
// ======================================================================
__global__ void k_conv3(const _Float16* __restrict__ cin, const _Float16* __restrict__ wt,
                        const float* __restrict__ bn, const float* __restrict__ pr,
                        const float* __restrict__ res, float* __restrict__ out,
                        _Float16* __restrict__ outh) {
  const int lane = threadIdx.x, hl = lane >> 4, m = lane & 15;
  const int bx = blockIdx.x;            // 8*64
  const int b = bx >> 6, i = bx & 63;
  const int jt = threadIdx.y & 3;       // j-tile
  const int nt = threadIdx.y >> 2;      // co-tile
  const int n  = nt * 16 + m;
  const v8h z8 = {};
  v8f acc = {};
#pragma unroll
  for (int s = 0; s < 9; ++s) {
    const int di = s / 3, dj = s % 3;
    const int ii = i + di - 1;
    const int jj = jt * 16 + m + dj - 1;
    const bool ok = ((unsigned)ii < 64u) && ((unsigned)jj < 64u);
    const int iic = ok ? ii : 0, jjc = ok ? jj : 0;
    const _Float16* ap = cin + (((size_t)b * HW + iic) * HW + jjc) * D_MODEL;
    const _Float16* bw = wt + (size_t)(s * 64 + n) * 64;
#pragma unroll
    for (int kc = 0; kc < D_MODEL; kc += 32) {
      v8h alo = *(const v8h*)&ap[kc + 8 * hl];
      v8h ahi = *(const v8h*)&ap[kc + 16 + 8 * hl];
      if (!ok) { alo = z8; ahi = z8; }
      const v16h a = __builtin_shufflevector(alo, ahi, 0,1,2,3,4,5,6,7,8,9,10,11,12,13,14,15);
      acc = wmma16(a, fragh(bw + kc + 8 * hl), acc);
    }
  }
  const float g = bn[n], be = bn[64 + n], mu = bn[128 + n], var = bn[192 + n];
  const float sc = g * rsqrtf(var + EPSBN);
  const float al = pr[0];
#pragma unroll
  for (int r = 0; r < 8; ++r) {
    const int jo = jt * 16 + r + 8 * hl;
    float v = (acc[r] - mu) * sc + be;
    v = (v >= 0.f) ? v : al * v;
    if (outh) {
      outh[(((size_t)b * HW + i) * HW + jo) * D_MODEL + n] = (_Float16)v;
    } else {
      const size_t o = (((size_t)b * D_MODEL + n) * HW + i) * HW + jo;
      out[o] = v + res[o];
    }
  }
}

// ======================================================================
extern "C" void kernel_launch(void* const* d_in, const int* in_sizes, int n_in,
                              void* d_out, int out_size, void* d_ws, size_t ws_size,
                              hipStream_t stream) {
  const float* x        = (const float*)d_in[0];
  const float* in_w     = (const float*)d_in[1];
  const float* c1d_w    = (const float*)d_in[2];
  const float* c1d_b    = (const float*)d_in[3];
  const float* xp_w     = (const float*)d_in[4];
  const float* dt_w     = (const float*)d_in[5];
  const float* dt_b     = (const float*)d_in[6];
  const float* A_log    = (const float*)d_in[7];
  const float* Dp       = (const float*)d_in[8];
  const float* out_w    = (const float*)d_in[9];
  const float* pr_ssm   = (const float*)d_in[10];
  const float* res_w    = (const float*)d_in[11];
  const float* res_bn   = (const float*)d_in[12];
  const float* conv1_w  = (const float*)d_in[13];
  const float* bn1      = (const float*)d_in[14];
  const float* pr1      = (const float*)d_in[15];
  const float* conv2_w  = (const float*)d_in[16];
  const float* bn2      = (const float*)d_in[17];
  const float* pr2      = (const float*)d_in[18];
  float* outp = (float*)d_out;

  // workspace layout
  char* wsb = (char*)d_ws;
  size_t off = 0;
  auto allocf = [&](size_t nfloat) { float* p = (float*)(wsb + off); off += nfloat * 4; return p; };
  auto alloch = [&](size_t nhalf)  { _Float16* p = (_Float16*)(wsb + off); off += ((nhalf * 2 + 15) & ~15ull); return p; };
  float*    xm    = allocf((size_t)NTOK * D_INNER);
  float*    zb    = allocf((size_t)NTOK * D_INNER);
  float*    xs    = allocf((size_t)NTOK * D_INNER);
  float*    yb    = allocf((size_t)NTOK * D_INNER);
  float*    xdbl  = allocf((size_t)NTOK * 36);
  float*    rblk  = allocf((size_t)NB * D_MODEL * HW * HW);
  _Float16* xt16  = alloch((size_t)NTOK * D_MODEL);
  _Float16* xs16  = alloch((size_t)NTOK * D_INNER);
  _Float16* cin16 = alloch((size_t)NB * HW * HW * D_MODEL);
  _Float16* c1h16 = alloch((size_t)NB * HW * HW * D_MODEL);
  _Float16* wt1   = alloch((size_t)9 * 64 * 64);
  _Float16* wt2   = alloch((size_t)9 * 64 * 64);
  _Float16* inw16 = alloch((size_t)256 * 64);
  _Float16* xpw16 = alloch((size_t)48 * 128);
  _Float16* ow16  = alloch((size_t)64 * 128);
  _Float16* rw16  = alloch((size_t)64 * 64);
  (void)ws_size; (void)in_sizes; (void)n_in; (void)out_size;

  // prep passes (tiny; weights converted to f16 exactly once)
  k_x2t<<<NB * 64, 256, 0, stream>>>(x, xt16);
  k_wprep<<<(9 * 64 * 64 + 255) / 256, 256, 0, stream>>>(conv1_w, wt1);
  k_wprep<<<(9 * 64 * 64 + 255) / 256, 256, 0, stream>>>(conv2_w, wt2);
  k_f32tof16<<<(256 * 64 + 255) / 256, 256, 0, stream>>>(in_w, inw16, 256 * 64);
  k_f32tof16<<<(64 * 128 + 255) / 256, 256, 0, stream>>>(out_w, ow16, 64 * 128);
  k_f32tof16<<<(64 * 64 + 255) / 256, 256, 0, stream>>>(res_w, rw16, 64 * 64);
  k_xpw<<<(48 * 128 + 255) / 256, 256, 0, stream>>>(xp_w, xpw16);

  // mamba branch
  k_inproj<<<dim3(NTOK / 16, 2), dim3(32, 8), 0, stream>>>(xt16, inw16, xm, zb);
  k_dwconv_silu<<<(NTOK * D_INNER + 255) / 256, 256, 0, stream>>>(xm, c1d_w, c1d_b, xs, xs16);
  k_xproj<<<NTOK / 16, dim3(32, 3), 0, stream>>>(xs16, xpw16, xdbl);
  k_scan<<<64, dim3(32, 8), 0, stream>>>(xdbl, xs, dt_w, dt_b, A_log, Dp, yb);
  k_gate_outproj<<<NTOK / 16, dim3(32, 4), 0, stream>>>(yb, zb, ow16, pr_ssm, cin16);

  // residual branch
  k_res<<<NB * 256, dim3(32, 4), 0, stream>>>(xt16, rw16, res_bn, rblk);

  // conv branch (+ fused final residual add)
  k_conv3<<<NB * 64, dim3(32, 16), 0, stream>>>(cin16, wt1, bn1, pr1, nullptr, nullptr, c1h16);
  k_conv3<<<NB * 64, dim3(32, 16), 0, stream>>>(c1h16, wt2, bn2, pr2, rblk, outp, nullptr);
}
